// Model_16028817948860
// MI455X (gfx1250) — matrix operand order repaired and checked by reading, not compiled, not run on hardware
//
#include <hip/hip_runtime.h>
#include <hip/hip_bf16.h>

typedef __attribute__((ext_vector_type(16))) _Float16 v16h;
typedef __attribute__((ext_vector_type(8)))  float    v8f;

static_assert(sizeof(v16h) == 32, "v16h size");
static_assert(sizeof(v8f) == 32, "v8f size");

// ---- CDNA5 async global->LDS staging (guarded; manual fallback) -----------
#if defined(__has_builtin)
#  if __has_builtin(__builtin_amdgcn_global_load_async_to_lds_b128)
#    define HAVE_ASYNC_LDS 1
#  endif
#  if __has_builtin(__builtin_amdgcn_s_wait_asynccnt)
#    define HAVE_WAIT_ASYNC 1
#  endif
#endif
#ifndef HAVE_ASYNC_LDS
#  define HAVE_ASYNC_LDS 0
#endif
#ifndef HAVE_WAIT_ASYNC
#  define HAVE_WAIT_ASYNC 0
#endif

#if HAVE_ASYNC_LDS
// builtin wants: (v4i AS1* src, v4i AS3* dst, imm offset, imm cpol)
typedef int v4i __attribute__((vector_size(16)));
typedef __attribute__((address_space(1))) v4i Gv4i;
typedef __attribute__((address_space(3))) v4i Lv4i;
#endif

__device__ __forceinline__ void wait_async0() {
#if HAVE_WAIT_ASYNC
    __builtin_amdgcn_s_wait_asynccnt(0);
#else
    asm volatile("s_wait_asynccnt 0x0" ::: "memory");
#endif
}

__device__ __forceinline__ v8f vzero() { v8f z = {}; return z; }

__device__ __forceinline__ v8f wmma_acc(v16h a, const _Float16* wcol, int kb2, v8f acc) {
    union { unsigned int u[8]; v16h h; } bf;
    const unsigned int* brow = (const unsigned int*)wcol;
#pragma unroll
    for (int v = 0; v < 8; ++v) bf.u[v] = brow[((v & 4) << 1) + (kb2 >> 1) + (v & 3)];
    return __builtin_amdgcn_wmma_f32_16x16x32_f16(false, a, false, bf.h, (short)0, acc, false, false);
}

// ---------------------------------------------------------------- utilities

// transpose + convert: src [K x 128] f32 -> dst [128 x K] f16
__global__ void k_convert_wT(const float* __restrict__ src, _Float16* __restrict__ dst, int K) {
    int i = blockIdx.x * 256 + threadIdx.x;
    if (i < K * 128) {
        int c = i / K, k = i - c * K;
        dst[(size_t)c * K + k] = (_Float16)src[(size_t)k * 128 + c];
    }
}

__global__ void k_zero(float* __restrict__ p, long n) {
    long i = (long)blockIdx.x * 256 + threadIdx.x;
    if (i < n) p[i] = 0.0f;
}

// ------------------------------------------------------------ edge features

__global__ void k_mesh_feat(const float* __restrict__ mesh_pos, const float* __restrict__ world_pos,
                            const int* __restrict__ s, const int* __restrict__ r,
                            const float* __restrict__ mean, const float* __restrict__ stdv,
                            float* __restrict__ feat, int E) {
    int e = blockIdx.x * 256 + threadIdx.x;
    if (e >= E) return;
    int a = s[e], b = r[e];
    float rx = mesh_pos[2*a]   - mesh_pos[2*b];
    float ry = mesh_pos[2*a+1] - mesh_pos[2*b+1];
    float rl = sqrtf(rx*rx + ry*ry);
    float wx = world_pos[2*a]   - world_pos[2*b];
    float wy = world_pos[2*a+1] - world_pos[2*b+1];
    float wl = sqrtf(wx*wx + wy*wy);
    float f[6] = {rx, ry, rl, wx, wy, wl};
#pragma unroll
    for (int j = 0; j < 6; ++j) feat[(size_t)e*6 + j] = (f[j] - mean[j]) / stdv[j];
}

__global__ void k_world_feat(const float* __restrict__ world_pos,
                             const int* __restrict__ s, const int* __restrict__ r,
                             const float* __restrict__ mean, const float* __restrict__ stdv,
                             float* __restrict__ feat, int E) {
    int e = blockIdx.x * 256 + threadIdx.x;
    if (e >= E) return;
    int a = r[e], b = s[e];
    float wx = world_pos[2*a]   - world_pos[2*b];
    float wy = world_pos[2*a+1] - world_pos[2*b+1];
    float wl = sqrtf(wx*wx + wy*wy);
    float f[3] = {wx, wy, wl};
#pragma unroll
    for (int j = 0; j < 3; ++j) feat[(size_t)e*3 + j] = (f[j] - mean[j]) / stdv[j];
}

// ------------------------------------------------ fused 3-layer WMMA MLP
//
// layer0: concat(A0[idx0],A1[idx1],A2[idx2]) [M x K0] @ W0 + b0, relu
// layer1: 128x128 @ W1 + b1, relu      (if W2t==null: write out here, done)
// layer2: 128x128 @ W2 + b2, layernorm(g,b), +res  -> out
// Weights are pre-transposed f16 [128 x K]. Hidden activations live in LDS.
// Tile: 64 rows x 128 cols; 8 wave32s; wave w: row-block (w&3), col-blocks
// 4*(w>>2)+{0..3}; v_wmma_f32_16x16x32_f16 over 32-wide K chunks.

__launch_bounds__(256)
__global__ void k_mlp3(
    const float* __restrict__ A0, const int* __restrict__ idx0, int d0,
    const float* __restrict__ A1, const int* __restrict__ idx1, int d1,
    const float* __restrict__ A2, const int* __restrict__ idx2, int d2,
    const _Float16* __restrict__ W0t, const float* __restrict__ b0,
    const _Float16* __restrict__ W1t, const float* __restrict__ b1,
    const _Float16* __restrict__ W2t, const float* __restrict__ b2,
    const float* __restrict__ ln_g, const float* __restrict__ ln_b,
    const float* __restrict__ res, float* __restrict__ out, int M)
{
    __shared__ __align__(16) _Float16 Asm[64 * 32];    //  4 KB layer-0 A staging
    __shared__ __align__(16) _Float16 Ah [64 * 128];   // 16 KB hidden acts (f16)
    __shared__ __align__(16) _Float16 Wsm[128 * 32];   //  8 KB W tile [col][k]
    __shared__ __align__(16) float    Osm[64 * 128];   // 32 KB f32 out tile
    __shared__ int ridx[3][64];

    const int tid  = threadIdx.x;
    const int row0 = blockIdx.x * 64;
    const int K0   = d0 + d1 + d2;

    if (tid < 64) {
        int grow = row0 + tid;
        bool ok = grow < M;
        ridx[0][tid] = ok ? (idx0 ? idx0[grow] : grow) : 0;
        ridx[1][tid] = ok ? (idx1 ? idx1[grow] : grow) : 0;
        ridx[2][tid] = ok ? (idx2 ? idx2[grow] : grow) : 0;
    }
    __syncthreads();

    const int lane = tid & 31, wv = tid >> 5;
    const int rb = wv & 3, cb0 = (wv >> 2) * 4;
    const int bl = lane & 15, kb2 = (lane >> 4) * 8;
    const int kq = (lane >> 4) * 4;
    const int arow = rb * 16 + (lane & 15);
    v8f acc0, acc1, acc2, acc3;

    // weight-chunk staging; [128][K] source, 16B-contiguous per column
    auto stageW_issue = [&](const _Float16* Wt, int K, int k0) -> bool {
#if HAVE_ASYNC_LDS
        if ((K & 31) == 0) {
            for (int p = tid; p < 512; p += 256) {
                const int c = p >> 2, j = (p & 3) * 8;
                __builtin_amdgcn_global_load_async_to_lds_b128(
                    (Gv4i*)(Wt + (size_t)c * K + k0 + j),
                    (Lv4i*)(&Wsm[c * 32 + j]), 0, 0);
            }
            return true;
        }
#endif
        for (int p = tid; p < 128 * 32; p += 256) {
            const int c = p >> 5, k = p & 31;
            const int kk = k0 + k;
            Wsm[c * 32 + k] = (kk < K) ? Wt[(size_t)c * K + kk] : (_Float16)0.0f;
        }
        return false;
    };

    // A fragment per documented CDNA5 16x32 f16 layout
    auto fragA = [&](const _Float16* base, int stride_h, int k0) -> v16h {
        union { unsigned int u[8]; v16h h; } af;
        const unsigned int* ar = (const unsigned int*)(base + (size_t)arow * stride_h + k0);
#pragma unroll
        for (int v = 0; v < 4; ++v) af.u[v] = ar[kq + v];
#pragma unroll
        for (int v = 0; v < 4; ++v) af.u[4 + v] = ar[8 + kq + v];
        return af.h;
    };

    auto mm4 = [&](v16h a) {
        acc0 = wmma_acc(a, &Wsm[((cb0 + 0) * 16 + bl) * 32], kb2, acc0);
        acc1 = wmma_acc(a, &Wsm[((cb0 + 1) * 16 + bl) * 32], kb2, acc1);
        acc2 = wmma_acc(a, &Wsm[((cb0 + 2) * 16 + bl) * 32], kb2, acc2);
        acc3 = wmma_acc(a, &Wsm[((cb0 + 3) * 16 + bl) * 32], kb2, acc3);
    };

    auto spill = [&]() {
        const int mh = 8 * (lane >> 4), nn = lane & 15;
#pragma unroll
        for (int v = 0; v < 8; ++v) {
            const int r = rb * 16 + mh + v;
            Osm[r * 128 + (cb0 + 0) * 16 + nn] = acc0[v];
            Osm[r * 128 + (cb0 + 1) * 16 + nn] = acc1[v];
            Osm[r * 128 + (cb0 + 2) * 16 + nn] = acc2[v];
            Osm[r * 128 + (cb0 + 3) * 16 + nn] = acc3[v];
        }
    };

    // ===================== layer 0 =====================
    acc0 = vzero(); acc1 = vzero(); acc2 = vzero(); acc3 = vzero();
    const int nk0 = (K0 + 31) >> 5;
    for (int kc = 0; kc < nk0; ++kc) {
        const int k0 = kc * 32;
        const bool as = stageW_issue(W0t, K0, k0);
        // chunk-uniform segment select (segments are 32-aligned or single-chunk)
        const float* As; const int* rix; int dS, kloc;
        if (k0 < d0)           { As = A0; rix = ridx[0]; dS = d0; kloc = k0; }
        else if (k0 < d0 + d1) { As = A1; rix = ridx[1]; dS = d1; kloc = k0 - d0; }
        else                   { As = A2; rix = ridx[2]; dS = d2; kloc = k0 - d0 - d1; }
        for (int p = tid; p < 64 * 16; p += 256) {
            const int r = p >> 4, kp = (p & 15) * 2;
            const int grow = row0 + r;
            float x0 = 0.0f, x1 = 0.0f;
            if (grow < M) {
                const float* rp = As + (size_t)rix[r] * dS + kloc;
                if (kloc + kp < dS)     x0 = rp[kp];
                if (kloc + kp + 1 < dS) x1 = rp[kp + 1];
            }
            union { _Float16 h[2]; unsigned int u; } pk;
            pk.h[0] = (_Float16)x0; pk.h[1] = (_Float16)x1;
            *(unsigned int*)&Asm[r * 32 + kp] = pk.u;
        }
        if (as) wait_async0();
        __syncthreads();
        mm4(fragA(Asm, 32, 0));
        __syncthreads();
    }
    spill();
    __syncthreads();
    for (int p = tid; p < 64 * 128; p += 256)
        Ah[p] = (_Float16)fmaxf(Osm[p] + b0[p & 127], 0.0f);
    __syncthreads();

    // ===================== layer 1 =====================
    acc0 = vzero(); acc1 = vzero(); acc2 = vzero(); acc3 = vzero();
    for (int kc = 0; kc < 4; ++kc) {
        const int k0 = kc * 32;
        const bool as = stageW_issue(W1t, 128, k0);
        if (as) wait_async0();
        __syncthreads();
        mm4(fragA(Ah, 128, k0));
        __syncthreads();
    }
    spill();
    __syncthreads();

    if (!W2t) {  // 2-layer mode (decoder trunk)
        for (int p = tid; p < 64 * 128; p += 256) {
            const int r = p >> 7, c = p & 127;
            const int grow = row0 + r;
            if (grow < M) out[(size_t)grow * 128 + c] = fmaxf(Osm[p] + b1[c], 0.0f);
        }
        return;
    }
    for (int p = tid; p < 64 * 128; p += 256)
        Ah[p] = (_Float16)fmaxf(Osm[p] + b1[p & 127], 0.0f);
    __syncthreads();

    // ===================== layer 2 + layernorm =====================
    acc0 = vzero(); acc1 = vzero(); acc2 = vzero(); acc3 = vzero();
    for (int kc = 0; kc < 4; ++kc) {
        const int k0 = kc * 32;
        const bool as = stageW_issue(W2t, 128, k0);
        if (as) wait_async0();
        __syncthreads();
        mm4(fragA(Ah, 128, k0));
        __syncthreads();
    }
    spill();
    __syncthreads();

    // layernorm: 4 lanes per row, shuffle reduction for row stats
    {
        const int r = tid >> 2, part = tid & 3;
        const int grow = row0 + r;
        float* rowp = &Osm[r * 128 + part * 32];
        const float* bp = b2 + part * 32;
        float s = 0.0f, ss = 0.0f;
#pragma unroll 8
        for (int c = 0; c < 32; ++c) {
            float x = rowp[c] + bp[c];
            rowp[c] = x; s += x; ss += x * x;
        }
        s  += __shfl_xor(s, 1);  s  += __shfl_xor(s, 2);
        ss += __shfl_xor(ss, 1); ss += __shfl_xor(ss, 2);
        const float mean = s * (1.0f / 128.0f);
        const float var  = ss * (1.0f / 128.0f) - mean * mean;
        const float inv  = rsqrtf(fmaxf(var, 0.0f) + 1e-5f);
        if (grow < M) {
            const int c0 = part * 32;
#pragma unroll 8
            for (int c = 0; c < 32; ++c) {
                float y = (rowp[c] - mean) * inv * ln_g[c0 + c] + ln_b[c0 + c];
                if (res) y += res[(size_t)grow * 128 + c0 + c];
                out[(size_t)grow * 128 + c0 + c] = y;
            }
        }
    }
}

// ---------------------------------------------------- scatter (segment_sum)
__global__ void k_scatter(const float* __restrict__ enew, const int* __restrict__ recv,
                          float* __restrict__ estate, float* __restrict__ agg, long n) {
    long i = (long)blockIdx.x * 256 + threadIdx.x;
    if (i >= n) return;
    const int e = (int)(i >> 7), c = (int)(i & 127);
    const float v = enew[i];
    estate[i] += v;
    atomicAdd(&agg[(size_t)recv[e] * 128 + c], v);
}

// ------------------------------------------------- decoder tail (128 -> 3)
__global__ void k_dec_final(const float* __restrict__ h, const float* __restrict__ w3,
                            const float* __restrict__ b3, const float* __restrict__ omean,
                            const float* __restrict__ ostd, const float* __restrict__ world_pos,
                            float* __restrict__ out, int N) {
    int i = blockIdx.x * 256 + threadIdx.x;
    if (i >= N) return;
    float a0 = 0.0f, a1 = 0.0f, a2 = 0.0f;
    const float* hr = h + (size_t)i * 128;
    for (int k = 0; k < 128; ++k) {
        float x = hr[k];
        a0 += x * w3[k * 3 + 0];
        a1 += x * w3[k * 3 + 1];
        a2 += x * w3[k * 3 + 2];
    }
    a0 = (a0 + b3[0]) * ostd[0] + omean[0];
    a1 = (a1 + b3[1]) * ostd[1] + omean[1];
    a2 = (a2 + b3[2]) * ostd[2] + omean[2];
    out[(size_t)i * 3 + 0] = world_pos[2*i]     + a0;
    out[(size_t)i * 3 + 1] = world_pos[2*i + 1] + a1;
    out[(size_t)i * 3 + 2] = a2;
}

// ------------------------------------------------------------------- host

static inline int cdiv(long a, long b) { return (int)((a + b - 1) / b); }

extern "C" void kernel_launch(void* const* d_in, const int* in_sizes, int n_in,
                              void* d_out, int out_size, void* d_ws, size_t ws_size,
                              hipStream_t stream) {
    const float* world_pos = (const float*)d_in[0];
    const float* mesh_pos  = (const float*)d_in[1];
    const float* node_oh   = (const float*)d_in[2];
    const int*   m_s       = (const int*)d_in[3];
    const int*   m_r       = (const int*)d_in[4];
    const int*   w_s       = (const int*)d_in[5];
    const int*   w_r       = (const int*)d_in[6];
    const int N  = in_sizes[0] / 2;
    const int Em = in_sizes[3];
    const int Ew = in_sizes[5];

    // params pytree flattening (dict keys sorted)
    const int PB = 7;
    auto P = [&](int i) -> const float* { return (const float*)d_in[PB + i]; };
    struct Mlp { const float *w0, *w1, *w2, *b0, *b1, *b2, *g, *lb; };
    auto lnMlp = [&](int base) -> Mlp {
        Mlp m;
        m.b0 = P(base); m.b1 = P(base + 1); m.b2 = P(base + 2);
        m.lb = P(base + 3); m.g = P(base + 4);
        m.w0 = P(base + 5); m.w1 = P(base + 6); m.w2 = P(base + 7);
        return m;
    };
    Mlp blkMesh[15], blkNode[15], blkWorld[15];
    for (int i = 0; i < 15; ++i) {
        blkMesh[i]  = lnMlp(24 * i);
        blkNode[i]  = lnMlp(24 * i + 8);
        blkWorld[i] = lnMlp(24 * i + 16);
    }
    Mlp dec;
    dec.b0 = P(360); dec.b1 = P(361); dec.b2 = P(362);
    dec.w0 = P(363); dec.w1 = P(364); dec.w2 = P(365);
    dec.g = nullptr; dec.lb = nullptr;
    Mlp encM = lnMlp(366), encN = lnMlp(374), encW = lnMlp(382);
    const float* mesh_mean  = P(390); const float* mesh_std  = P(391);
    const float* out_mean   = P(392); const float* out_std   = P(393);
    const float* world_mean = P(394); const float* world_std = P(395);
    (void)n_in; (void)out_size; (void)ws_size;

    // workspace layout
    char* wsb = (char*)d_ws;
    size_t off = 0;
    auto wsA = [&](size_t bytes) -> void* {
        void* p = wsb + off;
        off += (bytes + 255) & ~(size_t)255;
        return p;
    };
    _Float16* whb   = (_Float16*)wsA(8u * 1024u * 1024u);
    float* node_lat = (float*)wsA((size_t)N  * 128 * 4);
    float* me       = (float*)wsA((size_t)Em * 128 * 4);
    float* we       = (float*)wsA((size_t)Ew * 128 * 4);
    float* me_new   = (float*)wsA((size_t)Em * 128 * 4);
    float* we_new   = (float*)wsA((size_t)Ew * 128 * 4);
    float* agg_m    = (float*)wsA((size_t)N * 128 * 4);
    float* agg_w    = (float*)wsA((size_t)N * 128 * 4);
    float* dech     = (float*)wsA((size_t)N * 128 * 4);
    float* mfeat    = (float*)wsA((size_t)Em * 6 * 4);
    float* wfeat    = (float*)wsA((size_t)Ew * 3 * 4);

    // pre-convert + transpose weights to f16 [128 x K]
    size_t whOff = 0;
    auto conv = [&](const float* src, int K) -> const _Float16* {
        _Float16* dst = whb + whOff;
        long n = (long)K * 128;
        whOff += (size_t)((n + 7) & ~7L);
        k_convert_wT<<<cdiv(n, 256), 256, 0, stream>>>(src, dst, K);
        return dst;
    };
    struct MlpH { const _Float16 *w0, *w1, *w2; };
    auto convMlp = [&](const Mlp& m, int din) -> MlpH {
        MlpH h;
        h.w0 = conv(m.w0, din);
        h.w1 = conv(m.w1, 128);
        h.w2 = conv(m.w2, 128);
        return h;
    };
    MlpH encNh = convMlp(encN, 9);
    MlpH encMh = convMlp(encM, 6);
    MlpH encWh = convMlp(encW, 3);
    MlpH bMeshH[15], bNodeH[15], bWorldH[15];
    for (int i = 0; i < 15; ++i) {
        bMeshH[i]  = convMlp(blkMesh[i], 384);
        bNodeH[i]  = convMlp(blkNode[i], 384);
        bWorldH[i] = convMlp(blkWorld[i], 384);
    }
    MlpH decH;
    decH.w0 = conv(dec.w0, 128);
    decH.w1 = conv(dec.w1, 128);
    decH.w2 = nullptr;  // 128->3 tail stays f32

    auto mlp3 = [&](const float* A0, const int* i0, int d0,
                    const float* A1, const int* i1, int d1,
                    const float* A2, const int* i2, int d2,
                    const Mlp& m, const MlpH& mh,
                    const float* res, float* o, int M) {
        if (M <= 0) return;
        k_mlp3<<<cdiv(M, 64), 256, 0, stream>>>(A0, i0, d0, A1, i1, d1, A2, i2, d2,
                                                mh.w0, m.b0, mh.w1, m.b1, mh.w2, m.b2,
                                                m.g, m.lb, res, o, M);
    };

    // edge features
    if (Em > 0)
        k_mesh_feat<<<cdiv(Em, 256), 256, 0, stream>>>(mesh_pos, world_pos, m_s, m_r,
                                                       mesh_mean, mesh_std, mfeat, Em);
    if (Ew > 0)
        k_world_feat<<<cdiv(Ew, 256), 256, 0, stream>>>(world_pos, w_s, w_r,
                                                        world_mean, world_std, wfeat, Ew);

    // encoders
    mlp3(node_oh, nullptr, 9, nullptr, nullptr, 0, nullptr, nullptr, 0,
         encN, encNh, nullptr, node_lat, N);
    mlp3(mfeat, nullptr, 6, nullptr, nullptr, 0, nullptr, nullptr, 0,
         encM, encMh, nullptr, me, Em);
    mlp3(wfeat, nullptr, 3, nullptr, nullptr, 0, nullptr, nullptr, 0,
         encW, encWh, nullptr, we, Ew);

    // message-passing steps
    for (int s = 0; s < 15; ++s) {
        mlp3(node_lat, m_s, 128, node_lat, m_r, 128, me, nullptr, 128,
             blkMesh[s], bMeshH[s], nullptr, me_new, Em);
        mlp3(node_lat, w_s, 128, node_lat, w_r, 128, we, nullptr, 128,
             blkWorld[s], bWorldH[s], nullptr, we_new, Ew);
        k_zero<<<cdiv((long)N * 128, 256), 256, 0, stream>>>(agg_m, (long)N * 128);
        k_zero<<<cdiv((long)N * 128, 256), 256, 0, stream>>>(agg_w, (long)N * 128);
        if (Em > 0)
            k_scatter<<<cdiv((long)Em * 128, 256), 256, 0, stream>>>(me_new, m_r, me, agg_m, (long)Em * 128);
        if (Ew > 0)
            k_scatter<<<cdiv((long)Ew * 128, 256), 256, 0, stream>>>(we_new, w_r, we, agg_w, (long)Ew * 128);
        mlp3(node_lat, nullptr, 128, agg_m, nullptr, 128, agg_w, nullptr, 128,
             blkNode[s], bNodeH[s], node_lat, node_lat, N);
    }

    // decoder: 2-layer trunk via k_mlp3 (W2t == null), then f32 tail
    if (N > 0)
        k_mlp3<<<cdiv(N, 64), 256, 0, stream>>>(node_lat, nullptr, 128,
                                                nullptr, nullptr, 0, nullptr, nullptr, 0,
                                                decH.w0, dec.b0, decH.w1, dec.b1,
                                                nullptr, nullptr, nullptr, nullptr,
                                                nullptr, dech, N);
    k_dec_final<<<cdiv(N, 256), 256, 0, stream>>>(dech, dec.w2, dec.b2, out_mean, out_std,
                                                  world_pos, (float*)d_out, N);
}
